// LlamaSeerAttention_34832184770907
// MI455X (gfx1250) — compile-verified, matrix operand order checked by reading, not run on hardware
//
#include <hip/hip_runtime.h>

// ---------------------------------------------------------------------------
// Problem constants (fixed by the reference)
// ---------------------------------------------------------------------------
constexpr int S   = 2048;
constexpr int HID = 4096;
constexpr int NQ  = 32;
constexpr int NKV = 8;
constexpr int D   = 128;
constexpr int BLK = 64;
constexpr int GH  = 128;
constexpr int NB  = 32;     // S / BLK
constexpr int TOPK = 9;     // int((1 - sqrt(1-0.5)) * 32)

#define NEGF        (-3.4028235e38f)
#define INV_SQRT128 0.08838834764831845f

// ---------------------------------------------------------------------------
// gfx1250 async global->LDS path (probe via __has_builtin; clean fallback)
// ---------------------------------------------------------------------------
#if defined(__has_builtin)
#if __has_builtin(__builtin_amdgcn_global_load_async_to_lds_b128)
#define HAS_ASYNC_LDS 1
#endif
#if __has_builtin(__builtin_amdgcn_s_wait_asynccnt)
#define HAS_WAIT_ASYNC 1
#endif
#endif

typedef __attribute__((ext_vector_type(4))) int v4i;

__device__ inline void wait_async_all() {
#if defined(HAS_ASYNC_LDS)
#if defined(HAS_WAIT_ASYNC)
  __builtin_amdgcn_s_wait_asynccnt(0);
#else
  asm volatile("s_wait_asynccnt 0" ::: "memory");
#endif
#endif
}

// ---------------------------------------------------------------------------
// WMMA helpers (CDNA5 wave32, 16x16x32 f16 -> f32)
// ---------------------------------------------------------------------------
typedef __attribute__((ext_vector_type(16))) _Float16 v16h;
typedef __attribute__((ext_vector_type(8)))  _Float16 v8h;
typedef __attribute__((ext_vector_type(8)))  float    v8f;

__device__ inline v16h cat8(v8h lo, v8h hi) {
  return __builtin_shufflevector(lo, hi, 0,1,2,3,4,5,6,7,8,9,10,11,12,13,14,15);
}

__device__ inline v8f wmma16x16x32(v16h a, v16h b, v8f c) {
  return __builtin_amdgcn_wmma_f32_16x16x32_f16(
      /*neg_a=*/false, a, /*neg_b=*/false, b,
      /*c_mod=*/(short)0, c, /*reuse_a=*/false, /*reuse_b=*/false);
}

// One 16x32 f16 A/B fragment per the CDNA5 VGPR layout: lane owns row/col
// (lane&15); halves 0..7 = K kb..kb+7, halves 8..15 = K kb+16..kb+23.
// `row` already includes (lane&15); `kb8` already includes ((lane>>4)*8).
__device__ inline v16h load_frag(const _Float16* base, int stride, int row, int kb8) {
  const _Float16* p = base + (size_t)row * stride + kb8;
  v8h lo = *reinterpret_cast<const v8h*>(p);
  v8h hi = *reinterpret_cast<const v8h*>(p + 16);
  return cat8(lo, hi);
}

// ---------------------------------------------------------------------------
// Pure-f16 WMMA GEMM: C[M,N] = A[M,K] * WT[N,K]^T   (f16 in, fp32 out)
// Block tile 128x64, BK=64 halves, 256 threads = 8 waves (4x2), wave 32x32.
// A tile staged to LDS via async global->LDS b128 (raw copy, no cvt);
// B fragments stream K-contiguous straight from global WT (L2-resident).
// ---------------------------------------------------------------------------
constexpr int GEMM_BM = 128;
constexpr int GEMM_BN = 64;
constexpr int GEMM_BK = 64;   // halves per K step
constexpr int LDP     = 72;   // LDS row stride in halves (144B, 16B aligned)

__global__ __launch_bounds__(256)
void k_gemm_f16(const _Float16* __restrict__ A, const _Float16* __restrict__ WT,
                float* __restrict__ C, int M, int N, int K) {
  __shared__ _Float16 sA[GEMM_BM * LDP];

  const int tid  = threadIdx.x;
  const int lane = tid & 31;
  const int wave = tid >> 5;
  const int wm   = wave >> 1;          // 0..3
  const int wn   = wave & 1;           // 0..1
  const int m0   = blockIdx.y * GEMM_BM;
  const int n0   = blockIdx.x * GEMM_BN;
  const int laneR = lane & 15;
  const int laneK = (lane >> 4) * 8;

  v8f acc[2][2] = {};

  for (int k0 = 0; k0 < K; k0 += GEMM_BK) {
    // ---- stage A tile [128][64] halves: 1024 x 16B chunks, 4 per thread ----
#if defined(HAS_ASYNC_LDS)
    #pragma unroll
    for (int i = 0; i < 4; ++i) {
      int chunk = tid + 256 * i;              // 0..1023
      int r = chunk >> 3;                     // row 0..127
      int c = (chunk & 7) * 8;                // half col 0,8,..,56
      __builtin_amdgcn_global_load_async_to_lds_b128(
          (v4i*)(const_cast<_Float16*>(A) + (size_t)(m0 + r) * K + k0 + c),
          (v4i*)(&sA[r * LDP + c]), 0, 0);
    }
    wait_async_all();
#else
    uint4 stg[4];
    #pragma unroll
    for (int i = 0; i < 4; ++i) {
      int chunk = tid + 256 * i;
      int r = chunk >> 3, c = (chunk & 7) * 8;
      stg[i] = *reinterpret_cast<const uint4*>(A + (size_t)(m0 + r) * K + k0 + c);
    }
    #pragma unroll
    for (int i = 0; i < 4; ++i) {
      int chunk = tid + 256 * i;
      int r = chunk >> 3, c = (chunk & 7) * 8;
      *reinterpret_cast<uint4*>(&sA[r * LDP + c]) = stg[i];
    }
#endif
    // prefetch next K slice (global_prefetch_b8)
    if (k0 + GEMM_BK < K) {
      __builtin_prefetch(A + (size_t)(m0 + (tid >> 1)) * K + k0 + GEMM_BK, 0, 1);
      __builtin_prefetch(WT + (size_t)(n0 + (tid & 63)) * K + k0 + GEMM_BK, 0, 1);
    }
    __syncthreads();

    // ---- compute: 2x2 sub-tiles x 2 K-chunks -> 8 WMMA per wave ----
    v16h bf[2][2];
    #pragma unroll
    for (int nj = 0; nj < 2; ++nj)
      #pragma unroll
      for (int kc = 0; kc < 2; ++kc)
        bf[nj][kc] = load_frag(WT, K, n0 + wn * 32 + nj * 16 + laneR,
                               k0 + kc * 32 + laneK);
    #pragma unroll
    for (int mi = 0; mi < 2; ++mi) {
      v16h af[2];
      #pragma unroll
      for (int kc = 0; kc < 2; ++kc)
        af[kc] = load_frag(sA, LDP, wm * 32 + mi * 16 + laneR, kc * 32 + laneK);
      #pragma unroll
      for (int nj = 0; nj < 2; ++nj)
        #pragma unroll
        for (int kc = 0; kc < 2; ++kc)
          acc[mi][nj] = wmma16x16x32(af[kc], bf[nj][kc], acc[mi][nj]);
    }
    __syncthreads();
  }

  // ---- epilogue: C/D layout -> row = base + r + 8*(lane>>4), col = lane&15 ----
  const int rHi = (lane >> 4) * 8;
  #pragma unroll
  for (int mi = 0; mi < 2; ++mi)
    #pragma unroll
    for (int nj = 0; nj < 2; ++nj)
      #pragma unroll
      for (int r = 0; r < 8; ++r) {
        int row = m0 + wm * 32 + mi * 16 + rHi + r;
        int col = n0 + wn * 32 + nj * 16 + laneR;
        C[(size_t)row * N + col] = acc[mi][nj][r];
      }
}

// ---------------------------------------------------------------------------
// One-time weight transpose+convert: W[K][N] fp32 -> WT[N][K] f16 (L2-resident)
// ---------------------------------------------------------------------------
__global__ __launch_bounds__(256)
void k_transpose_f16(const float* __restrict__ W, _Float16* __restrict__ WT,
                     int K, int N) {
  __shared__ float t[32][33];
  const int kb = blockIdx.y * 32, nb = blockIdx.x * 32;
  const int tx = threadIdx.x & 31, ty = threadIdx.x >> 5;   // 8 rows of 32
  #pragma unroll
  for (int i = 0; i < 4; ++i) {
    int k = ty + i * 8;
    t[k][tx] = W[(size_t)(kb + k) * N + nb + tx];
  }
  __syncthreads();
  #pragma unroll
  for (int i = 0; i < 4; ++i) {
    int n = ty + i * 8;
    WT[(size_t)(nb + n) * K + kb + tx] = (_Float16)t[tx][n];
  }
}

// fp32 -> f16 elementwise (same layout)
__global__ void k_cvt_f16(const float* __restrict__ src, _Float16* __restrict__ dst,
                          int n) {
  int i = blockIdx.x * blockDim.x + threadIdx.x;
  if (i < n) dst[i] = (_Float16)src[i];
}

// ---------------------------------------------------------------------------
// RoPE (theta=10000, head_dim=128) fp32 [s][H*128] -> f16 head-major [h][s][128]
// ---------------------------------------------------------------------------
__global__ void k_rope_pack(const float* __restrict__ src, _Float16* __restrict__ dst, int H) {
  const int s = blockIdx.x, h = blockIdx.y, i = threadIdx.x;   // i in [0,64)
  const float inv = __powf(10000.f, -(float)(2 * i) / 128.f);
  const float ang = (float)s * inv;
  const float c = __cosf(ang), sn = __sinf(ang);
  const float x1 = src[(size_t)s * (H * D) + h * D + i];
  const float x2 = src[(size_t)s * (H * D) + h * D + i + 64];
  _Float16* o = dst + ((size_t)h * S + s) * D;
  o[i]      = (_Float16)(x1 * c - x2 * sn);
  o[i + 64] = (_Float16)(x2 * c + x1 * sn);
}

__global__ void k_cvt_v(const float* __restrict__ v, _Float16* __restrict__ vh) {
  const int s = blockIdx.x, h = blockIdx.y, d = threadIdx.x;
  vh[((size_t)h * S + s) * D + d] = (_Float16)v[(size_t)s * (NKV * D) + h * D + d];
}

// ---------------------------------------------------------------------------
// Gate path (tiny): block pooling, projection, block-RoPE, logits
// ---------------------------------------------------------------------------
__global__ void k_gate_poolq(const float* __restrict__ q, float* __restrict__ qpool) {
  const int nb = blockIdx.x, h = blockIdx.y, d = threadIdx.x;
  float s = 0.f;
  for (int j = 0; j < BLK; ++j) s += q[(size_t)(nb * BLK + j) * HID + h * D + d];
  qpool[(size_t)(h * NB + nb) * D + d] = s * (1.f / BLK);
}

__global__ void k_gate_poolk(const float* __restrict__ k, float* __restrict__ kpool) {
  const int nb = blockIdx.x, h = blockIdx.y, d = threadIdx.x;
  float mx = NEGF, mn = 3.4028235e38f, s = 0.f;
  for (int j = 0; j < BLK; ++j) {
    float x = k[(size_t)(nb * BLK + j) * (NKV * D) + h * D + d];
    mx = fmaxf(mx, x); mn = fminf(mn, x); s += x;
  }
  float* base = kpool + (size_t)(h * NB + nb) * (3 * D);
  base[d]         = mx;
  base[D + d]     = mn;
  base[2 * D + d] = s * (1.f / BLK);
}

__global__ void k_gate_proj(const float* __restrict__ pool, const float* __restrict__ wgt,
                            float* __restrict__ out, int Kdim) {
  __shared__ float tmp[GH];
  const int nb = blockIdx.x, h = blockIdx.y, g = threadIdx.x;
  const float* pv = pool + (size_t)(h * NB + nb) * Kdim;
  float a = 0.f;
  for (int d = 0; d < Kdim; ++d) a += pv[d] * wgt[(size_t)d * GH + g];
  tmp[g] = a;
  __syncthreads();
  const int i = g & 63;
  const float inv = __powf(10000.f, -(float)(2 * i) / (float)GH);
  const float ang = (float)nb * inv;
  const float c = __cosf(ang), sn = __sinf(ang);
  const float o = (g < 64) ? (tmp[g] * c - tmp[g + 64] * sn)
                           : (tmp[g] * c + tmp[g - 64] * sn);
  out[(size_t)(h * NB + nb) * GH + g] = o;
}

__global__ void k_gate_logits(const float* __restrict__ qg, const float* __restrict__ kg,
                              float* __restrict__ pred) {
  const int qb = blockIdx.x, h = blockIdx.y, kb = threadIdx.x;
  float v = NEGF;
  if (kb <= qb) {
    const float* a = qg + (size_t)(h * NB + qb) * GH;
    const float* b = kg + (size_t)((h >> 2) * NB + kb) * GH;   // repeat_kv
    float s = 0.f;
    for (int d = 0; d < GH; ++d) s += a[d] * b[d];
    v = s * INV_SQRT128;
  }
  pred[(size_t)(h * NB + qb) * NB + kb] = v;
}

__global__ void k_topk_mask(const float* __restrict__ pred, int* __restrict__ msk) {
  const int t = blockIdx.x * blockDim.x + threadIdx.x;
  if (t >= NQ * NB) return;
  const int h = t >> 5, qb = t & 31;
  const float* row = pred + (size_t)(h * NB + qb) * NB;
  float vals[NB];
  bool  sel[NB];
  for (int j = 0; j < NB; ++j) { vals[j] = row[j]; sel[j] = false; }
  for (int it = 0; it < TOPK; ++it) {
    float best = -__builtin_inff(); int bj = 0;
    for (int j = 0; j < NB; ++j)
      if (vals[j] > best) { best = vals[j]; bj = j; }
    sel[bj] = true; vals[bj] = -__builtin_inff();
  }
  for (int j = 0; j < NB; ++j)
    msk[(size_t)(h * NB + qb) * NB + j] = ((sel[j] || j == qb) && j <= qb) ? 1 : 0;
}

// ---------------------------------------------------------------------------
// pooling_gt: dense causal attention, streaming online softmax per q-block,
// per-row per-k-block partial sums.  One WG per (head h, q-block qb).
// ---------------------------------------------------------------------------
__global__ __launch_bounds__(256)
void k_pool_attn(const _Float16* __restrict__ qh, const _Float16* __restrict__ kh,
                 float* __restrict__ pool_out) {
  __shared__ float sc[64 * 65];
  __shared__ float srow[64 * 33];
  __shared__ float sm[64], sl[64];

  const int h = blockIdx.x, qb = blockIdx.y;
  const int tid = threadIdx.x, lane = tid & 31, wave = tid >> 5;
  const int mi = wave >> 1;
  const int njBase = (wave & 1) * 2;
  const int laneR = lane & 15;
  const int laneK = (lane >> 4) * 8;
  const int rHi = (lane >> 4) * 8;

  const _Float16* Q  = qh + ((size_t)h * S + qb * BLK) * D;
  const _Float16* Kb = kh + ((size_t)(h >> 2) * S) * D;

  v16h qf[4];
  #pragma unroll
  for (int kc = 0; kc < 4; ++kc)
    qf[kc] = load_frag(Q, D, mi * 16 + laneR, kc * 32 + laneK);

  if (tid < 64) {
    sm[tid] = NEGF; sl[tid] = 0.f;
    for (int j = 0; j < NB; ++j) srow[tid * 33 + j] = 0.f;
  }
  __syncthreads();

  for (int kb = 0; kb <= qb; ++kb) {
    v8f acc[2] = {};
    #pragma unroll
    for (int nj = 0; nj < 2; ++nj) {
      const int colTok = kb * BLK + (njBase + nj) * 16 + laneR;
      #pragma unroll
      for (int kc = 0; kc < 4; ++kc) {
        v16h bf = load_frag(Kb, D, colTok, kc * 32 + laneK);
        acc[nj] = wmma16x16x32(qf[kc], bf, acc[nj]);
      }
    }
    #pragma unroll
    for (int nj = 0; nj < 2; ++nj)
      #pragma unroll
      for (int r = 0; r < 8; ++r)
        sc[(mi * 16 + rHi + r) * 65 + (njBase + nj) * 16 + laneR] =
            acc[nj][r] * INV_SQRT128;
    __syncthreads();

    if (tid < 64) {
      const int r = tid;
      const float mOld = sm[r];
      float mx = mOld;
      for (int j = 0; j < 64; ++j) {
        bool ok = (kb < qb) || (j <= r);
        if (ok) mx = fmaxf(mx, sc[r * 65 + j]);
      }
      const float scl = __expf(mOld - mx);
      float t = 0.f;
      for (int j = 0; j < 64; ++j) {
        bool ok = (kb < qb) || (j <= r);
        if (ok) t += __expf(sc[r * 65 + j] - mx);
      }
      for (int j = 0; j < NB; ++j) srow[r * 33 + j] *= scl;
      srow[r * 33 + kb] = t;
      sl[r] = sl[r] * scl + t;
      sm[r] = mx;
    }
    __syncthreads();
  }

  if (tid < NB) {
    float a = 0.f;
    if (tid <= qb) {
      for (int r = 0; r < 64; ++r) a += srow[r * 33 + tid] / sl[r];
      a *= (1.f / BLK);
    }
    pool_out[(size_t)(h * NB + qb) * NB + tid] = a;
  }
}

// ---------------------------------------------------------------------------
// Block-sparse flash attention; visits only masked kblocks.  Output written
// directly as f16 [s][h*128+d] for the final WMMA GEMM.
// ---------------------------------------------------------------------------
__global__ __launch_bounds__(256)
void k_sparse_attn(const _Float16* __restrict__ qh, const _Float16* __restrict__ kh,
                   const _Float16* __restrict__ vh, const int* __restrict__ mask,
                   _Float16* __restrict__ attn) {
  __shared__ float    sc[64 * 65];
  __shared__ _Float16 sp[64 * 72];
  __shared__ _Float16 svt[128 * 72];
  __shared__ float sm[64], sl[64], sscale[64];

  const int h = blockIdx.x, qb = blockIdx.y;
  const int tid = threadIdx.x, lane = tid & 31, wave = tid >> 5;
  const int mi = wave & 3;
  const int grp = wave >> 2;
  const int njBase = grp * 2;
  const int djBase = grp * 4;
  const int laneR = lane & 15;
  const int laneK = (lane >> 4) * 8;
  const int rHi = (lane >> 4) * 8;

  const int* mrow = mask + (size_t)(h * NB + qb) * NB;
  const _Float16* Q  = qh + ((size_t)h * S + qb * BLK) * D;
  const _Float16* Kb = kh + ((size_t)(h >> 2) * S) * D;
  const _Float16* Vb = vh + ((size_t)(h >> 2) * S) * D;

  v16h qf[4];
  #pragma unroll
  for (int kc = 0; kc < 4; ++kc)
    qf[kc] = load_frag(Q, D, mi * 16 + laneR, kc * 32 + laneK);

  v8f oacc[4] = {};
  if (tid < 64) { sm[tid] = NEGF; sl[tid] = 0.f; }
  __syncthreads();

  for (int kb = 0; kb <= qb; ++kb) {
    if (!mrow[kb]) continue;

    #pragma unroll 4
    for (int i = 0; i < 32; ++i) {
      int e = tid + 256 * i;
      int r = e >> 7, d = e & 127;
      svt[d * 72 + r] = Vb[(size_t)(kb * BLK + r) * D + d];
    }

    v8f acc[2] = {};
    #pragma unroll
    for (int nj = 0; nj < 2; ++nj) {
      const int colTok = kb * BLK + (njBase + nj) * 16 + laneR;
      #pragma unroll
      for (int kc = 0; kc < 4; ++kc) {
        v16h bf = load_frag(Kb, D, colTok, kc * 32 + laneK);
        acc[nj] = wmma16x16x32(qf[kc], bf, acc[nj]);
      }
    }
    #pragma unroll
    for (int nj = 0; nj < 2; ++nj)
      #pragma unroll
      for (int r = 0; r < 8; ++r)
        sc[(mi * 16 + rHi + r) * 65 + (njBase + nj) * 16 + laneR] =
            acc[nj][r] * INV_SQRT128;
    __syncthreads();

    if (tid < 64) {
      const int r = tid;
      const float mOld = sm[r];
      float mx = mOld;
      for (int j = 0; j < 64; ++j) {
        bool ok = (kb < qb) || (j <= r);
        if (ok) mx = fmaxf(mx, sc[r * 65 + j]);
      }
      const float scl = __expf(mOld - mx);
      float t = 0.f;
      for (int j = 0; j < 64; ++j) {
        bool ok = (kb < qb) || (j <= r);
        float e = ok ? __expf(sc[r * 65 + j] - mx) : 0.f;
        sp[r * 72 + j] = (_Float16)e;
        t += e;
      }
      sl[r] = sl[r] * scl + t;
      sm[r] = mx;
      sscale[r] = scl;
    }
    __syncthreads();

    float rs[8];
    #pragma unroll
    for (int r = 0; r < 8; ++r) rs[r] = sscale[mi * 16 + rHi + r];
    #pragma unroll
    for (int dj = 0; dj < 4; ++dj)
      #pragma unroll
      for (int r = 0; r < 8; ++r) oacc[dj][r] *= rs[r];

    v16h pf[2];
    #pragma unroll
    for (int kc2 = 0; kc2 < 2; ++kc2)
      pf[kc2] = load_frag(sp, 72, mi * 16 + laneR, kc2 * 32 + laneK);
    #pragma unroll
    for (int dj = 0; dj < 4; ++dj) {
      #pragma unroll
      for (int kc2 = 0; kc2 < 2; ++kc2) {
        v16h vf = load_frag(svt, 72, (djBase + dj) * 16 + laneR, kc2 * 32 + laneK);
        oacc[dj] = wmma16x16x32(pf[kc2], vf, oacc[dj]);
      }
    }
    __syncthreads();
  }

  float li[8];
  #pragma unroll
  for (int r = 0; r < 8; ++r) li[r] = 1.f / sl[mi * 16 + rHi + r];
  #pragma unroll
  for (int dj = 0; dj < 4; ++dj)
    #pragma unroll
    for (int r = 0; r < 8; ++r) {
      int row = qb * BLK + mi * 16 + rHi + r;
      int col = h * D + (djBase + dj) * 16 + laneR;
      attn[(size_t)row * HID + col] = (_Float16)(oacc[dj][r] * li[r]);
    }
}

// ---------------------------------------------------------------------------
// Workspace layout (bytes)
// ---------------------------------------------------------------------------
constexpr size_t OFF_HH  = 0;                                    // f16 [S][HID]
constexpr size_t OFF_Q   = OFF_HH  + (size_t)S * HID * 2;        // f32 [S][4096]
constexpr size_t OFF_K   = OFF_Q   + (size_t)S * HID * 4;        // f32 [S][1024]
constexpr size_t OFF_V   = OFF_K   + (size_t)S * NKV * D * 4;
constexpr size_t OFF_ATTH= OFF_V   + (size_t)S * NKV * D * 4;    // f16 [S][HID]
constexpr size_t OFF_QH  = OFF_ATTH+ (size_t)S * HID * 2;        // f16 [NQ][S][D]
constexpr size_t OFF_KH  = OFF_QH  + (size_t)S * HID * 2;
constexpr size_t OFF_VH  = OFF_KH  + (size_t)S * NKV * D * 2;
constexpr size_t OFF_WQT = OFF_VH  + (size_t)S * NKV * D * 2;    // f16 [4096][4096]
constexpr size_t OFF_WKT = OFF_WQT + (size_t)HID * HID * 2;      // f16 [1024][4096]
constexpr size_t OFF_WVT = OFF_WKT + (size_t)NKV * D * HID * 2;
constexpr size_t OFF_WOT = OFF_WVT + (size_t)NKV * D * HID * 2;  // f16 [4096][4096]
constexpr size_t OFF_QP  = OFF_WOT + (size_t)HID * HID * 2;
constexpr size_t OFF_KP  = OFF_QP  + (size_t)NQ * NB * D * 4;
constexpr size_t OFF_QG  = OFF_KP  + (size_t)NKV * NB * 3 * D * 4;
constexpr size_t OFF_KG  = OFF_QG  + (size_t)NQ * NB * GH * 4;
constexpr size_t OFF_MSK = OFF_KG  + (size_t)NKV * NB * GH * 4;

extern "C" void kernel_launch(void* const* d_in, const int* in_sizes, int n_in,
                              void* d_out, int out_size, void* d_ws, size_t ws_size,
                              hipStream_t stream) {
  (void)in_sizes; (void)n_in; (void)out_size; (void)ws_size;

  const float* hs  = (const float*)d_in[0];
  const float* wq  = (const float*)d_in[1];
  const float* wk  = (const float*)d_in[2];
  const float* wv  = (const float*)d_in[3];
  const float* wo  = (const float*)d_in[4];
  const float* gwq = (const float*)d_in[5];
  const float* gwk = (const float*)d_in[6];

  char* ws = (char*)d_ws;
  _Float16* hh   = (_Float16*)(ws + OFF_HH);
  float*    q    = (float*)(ws + OFF_Q);
  float*    k    = (float*)(ws + OFF_K);
  float*    v    = (float*)(ws + OFF_V);
  _Float16* atth = (_Float16*)(ws + OFF_ATTH);
  _Float16* qhh  = (_Float16*)(ws + OFF_QH);
  _Float16* khh  = (_Float16*)(ws + OFF_KH);
  _Float16* vhh  = (_Float16*)(ws + OFF_VH);
  _Float16* wqT  = (_Float16*)(ws + OFF_WQT);
  _Float16* wkT  = (_Float16*)(ws + OFF_WKT);
  _Float16* wvT  = (_Float16*)(ws + OFF_WVT);
  _Float16* woT  = (_Float16*)(ws + OFF_WOT);
  float*    qp   = (float*)(ws + OFF_QP);
  float*    kp   = (float*)(ws + OFF_KP);
  float*    qg   = (float*)(ws + OFF_QG);
  float*    kg   = (float*)(ws + OFF_KG);
  int*      msk  = (int*)(ws + OFF_MSK);

  float* out     = (float*)d_out;                    // [S][HID]
  float* pool_gt = out + (size_t)S * HID;            // [NQ][NB][NB]
  float* pred    = pool_gt + (size_t)NQ * NB * NB;   // [NQ][NB][NB]

  // 0) one-time f16 conversions / weight transposes (L2-resident afterwards)
  k_cvt_f16<<<dim3((S * HID + 255) / 256), 256, 0, stream>>>(hs, hh, S * HID);
  k_transpose_f16<<<dim3(HID / 32, HID / 32), 256, 0, stream>>>(wq, wqT, HID, HID);
  k_transpose_f16<<<dim3((NKV * D) / 32, HID / 32), 256, 0, stream>>>(wk, wkT, HID, NKV * D);
  k_transpose_f16<<<dim3((NKV * D) / 32, HID / 32), 256, 0, stream>>>(wv, wvT, HID, NKV * D);
  k_transpose_f16<<<dim3(HID / 32, HID / 32), 256, 0, stream>>>(wo, woT, HID, HID);

  // 1) QKV projections (pure-f16 WMMA GEMMs, async-LDS staged A)
  k_gemm_f16<<<dim3(HID / GEMM_BN, S / GEMM_BM), 256, 0, stream>>>(hh, wqT, q, S, HID, HID);
  k_gemm_f16<<<dim3((NKV * D) / GEMM_BN, S / GEMM_BM), 256, 0, stream>>>(hh, wkT, k, S, NKV * D, HID);
  k_gemm_f16<<<dim3((NKV * D) / GEMM_BN, S / GEMM_BM), 256, 0, stream>>>(hh, wvT, v, S, NKV * D, HID);

  // 2) Gate pooling on pre-RoPE q/k
  k_gate_poolq<<<dim3(NB, NQ), D, 0, stream>>>(q, qp);
  k_gate_poolk<<<dim3(NB, NKV), D, 0, stream>>>(k, kp);

  // 3) RoPE + f16 head-major repack
  k_rope_pack<<<dim3(S, NQ),  64, 0, stream>>>(q, qhh, NQ);
  k_rope_pack<<<dim3(S, NKV), 64, 0, stream>>>(k, khh, NKV);
  k_cvt_v<<<dim3(S, NKV), D, 0, stream>>>(v, vhh);

  // 4) Gate projections + block-RoPE + logits (predict_mask)
  k_gate_proj<<<dim3(NB, NQ),  GH, 0, stream>>>(qp, gwq, qg, D);
  k_gate_proj<<<dim3(NB, NKV), GH, 0, stream>>>(kp, gwk, kg, 3 * D);
  k_gate_logits<<<dim3(NB, NQ), NB, 0, stream>>>(qg, kg, pred);

  // 5) Top-k block routing mask
  k_topk_mask<<<dim3((NQ * NB + 255) / 256), 256, 0, stream>>>(pred, msk);

  // 6) pooling ground truth (streaming dense attention)
  k_pool_attn<<<dim3(NQ, NB), 256, 0, stream>>>(qhh, khh, pool_gt);

  // 7) block-sparse attention (f16 output)
  k_sparse_attn<<<dim3(NQ, NB), 256, 0, stream>>>(qhh, khh, vhh, msk, atth);

  // 8) output projection
  k_gemm_f16<<<dim3(HID / GEMM_BN, S / GEMM_BM), 256, 0, stream>>>(atth, woT, out, S, HID, HID);
}
